// MultiheadAttention_45483703665088
// MI455X (gfx1250) — compile-verified, matrix operand order checked by reading, not run on hardware
//
#include <hip/hip_runtime.h>
#include <hip/hip_bf16.h>
#include <math.h>

#define BB 4
#define SS 2048
#define DD 1024
#define HH 16
#define HD 64

typedef __attribute__((ext_vector_type(16))) _Float16     v16h;
typedef __attribute__((ext_vector_type(8)))  float        v8f;
typedef __attribute__((ext_vector_type(4)))  unsigned int v4u;
typedef __attribute__((ext_vector_type(8)))  int          v8i;
typedef __attribute__((ext_vector_type(4)))  int          v4i;

#if defined(__AMDGCN__) && __has_builtin(__builtin_amdgcn_tensor_load_to_lds) && \
    __has_builtin(__builtin_amdgcn_s_wait_tensorcnt)
#define USE_TDM 1
#endif

// LDS layout for the attention kernel (bytes)
#define LDS_SC    0                                  // 16*SS f32 raw scores
#define LDS_PS    (16 * SS * 4)                      // 16*SS f16 probabilities
#define LDS_KBUF0 (LDS_PS + 16 * SS * 2)             // 16*HD f16 K tile, buf 0
#define LDS_KBUF1 (LDS_KBUF0 + 16 * HD * 2)          // buf 1
#define LDS_VBUF0 (LDS_KBUF1 + 16 * HD * 2)          // 32*HD f16 V tile, buf 0
#define LDS_VBUF1 (LDS_VBUF0 + 32 * HD * 2)          // buf 1
#define LDS_STATS (LDS_VBUF1 + 32 * HD * 2)          // 32 f32 row stats
#define LDS_TOTAL (LDS_STATS + 32 * 4)

// Starting K index of VGPR pair i for 16-bit WMMA fragments (ISA 7.12.2):
// lanes 0-15 (hi=0): VGPR0..3 -> K=0,2,4,6 ; VGPR4..7 -> K=16,18,20,22
// lanes 16-31 (hi=1): +8
__device__ __forceinline__ int frag_k(int i, int hi) {
  int base = (i < 4) ? (2 * i) : (16 + 2 * (i - 4));
  return base + hi * 8;
}

// A-fragment (16 rows x 32 K) or B-fragment whose N-column is contiguous along
// K in memory: f16 source, row-major, leading dim ld.
__device__ __forceinline__ v16h load_frag_f16(const _Float16* src, int ld) {
  const int lane = threadIdx.x & 31;
  const int r = lane & 15;
  const int hi = lane >> 4;
  v16h f;
#pragma unroll
  for (int i = 0; i < 8; ++i) {
    const int k = frag_k(i, hi);
    f[2 * i]     = src[(size_t)r * ld + k];
    f[2 * i + 1] = src[(size_t)r * ld + k + 1];
  }
  return f;
}

// Same but converting from f32 source on the fly.
__device__ __forceinline__ v16h load_frag_f32(const float* src, int ld) {
  const int lane = threadIdx.x & 31;
  const int r = lane & 15;
  const int hi = lane >> 4;
  v16h f;
#pragma unroll
  for (int i = 0; i < 8; ++i) {
    const int k = frag_k(i, hi);
    const float2 p = *(const float2*)(src + (size_t)r * ld + k);
    f[2 * i]     = (_Float16)p.x;
    f[2 * i + 1] = (_Float16)p.y;
  }
  return f;
}

// B-fragment where B[k][n] = src[k*ld + n] (K strided in memory), n = lane&15.
__device__ __forceinline__ v16h load_b_strided_f16(const _Float16* src, int ld) {
  const int lane = threadIdx.x & 31;
  const int c = lane & 15;
  const int hi = lane >> 4;
  v16h f;
#pragma unroll
  for (int i = 0; i < 8; ++i) {
    const int k = frag_k(i, hi);
    f[2 * i]     = src[(size_t)k * ld + c];
    f[2 * i + 1] = src[((size_t)k + 1) * ld + c];
  }
  return f;
}

#ifdef USE_TDM
// Issue a TDM DMA of `nelem` contiguous f16 elements from global to LDS.
// D# per ISA ch.8: group0 = {count=1, lds_addr, global_addr[56:0], type=2},
// group1 = {data_size=2B, tensor_dim0=nelem, tensor_dim1=1, tile_dim0=nelem,
//           tile_dim1/2=0 (1-D tile), tensor_dim0_stride=nelem}.
__device__ __forceinline__ void tdm_load_f16(unsigned lds_off, const _Float16* gsrc,
                                             int nelem) {
  const unsigned long long ga = (unsigned long long)(size_t)gsrc;
  v4u g0;
  g0[0] = 1u;                                             // count=1, user mode
  g0[1] = lds_off;                                        // lds_addr (bytes)
  g0[2] = (unsigned)ga;                                   // global_addr[31:0]
  g0[3] = (unsigned)((ga >> 32) & 0x01FFFFFFu) | (2u << 30);  // addr[56:32]|type=2
  v8i g1;
  g1[0] = (int)(1u << 16);                                // data_size = 2 bytes
  g1[1] = (int)(((unsigned)nelem & 0xffffu) << 16);       // tensor_dim0[15:0]
  g1[2] = (int)(((unsigned)nelem >> 16) | (1u << 16));    // dim0 hi | tensor_dim1=1
  g1[3] = (int)(((unsigned)nelem) << 16);                 // tile_dim0
  g1[4] = 0;                                              // tile_dim1/2 = 0 (1-D)
  g1[5] = nelem;                                          // tensor_dim0_stride
  g1[6] = 0;
  g1[7] = 0;
  v4i z = {};
#if __clang_major__ >= 23
  v8i z8 = {};
  __builtin_amdgcn_tensor_load_to_lds(g0, g1, z, z, z8, 0);
#else
  __builtin_amdgcn_tensor_load_to_lds(g0, g1, z, z, 0);
#endif
}
#endif

// ---------------------------------------------------------------------------
// Kernel 1: Y = (X @ W^T + bias) * scale, output f16 in head-split layout
// [B, H, S, HD]. One 16x16 output tile per wave; K loop = 32 WMMAs.
// ---------------------------------------------------------------------------
__global__ void proj_qkv_kernel(const float* __restrict__ X, const float* __restrict__ W,
                                const float* __restrict__ bias, _Float16* __restrict__ Y,
                                float scale) {
  const int wave = blockIdx.x * (blockDim.x >> 5) + (threadIdx.x >> 5);
  const int nT = DD / 16;                // 64 tiles along N
  const int mt = wave / nT;
  const int nt = wave % nT;
  const int lane = threadIdx.x & 31;
  const float* Xt = X + (size_t)mt * 16 * DD;
  const float* Wt = W + (size_t)nt * 16 * DD;   // row n of W is contiguous along k
  v8f acc = {};
#pragma unroll 4
  for (int kt = 0; kt < DD; kt += 32) {
    v16h a = load_frag_f32(Xt + kt, DD);
    v16h b = load_frag_f32(Wt + kt, DD);
    acc = __builtin_amdgcn_wmma_f32_16x16x32_f16(false, a, false, b, (short)0, acc,
                                                 false, false);
  }
  const int hi = lane >> 4;
  const int n = nt * 16 + (lane & 15);
  const int h = n / HD, hd = n % HD;
  const float bn = bias[n];
#pragma unroll
  for (int j = 0; j < 8; ++j) {
    const int row = mt * 16 + j + hi * 8;   // global row in [0, B*S)
    const int b_ = row / SS, s = row % SS;
    const float v = (acc[j] + bn) * scale;
    Y[(((size_t)b_ * HH + h) * SS + s) * HD + hd] = (_Float16)v;
  }
}

// ---------------------------------------------------------------------------
// Kernel 2: fused scores + bias + causal mask + softmax + P@V.
// One wave per (b, h, 16-query tile). Raw scores live in LDS (16 x 2048 f32),
// probabilities written ONCE to global attn output; f16 P kept in LDS for the
// P@V WMMA pass. K/V tiles are DMA'd into LDS by the Tensor Data Mover with
// TENSORcnt double-buffering when available. Fully-masked key tiles skipped.
// ---------------------------------------------------------------------------
__global__ void attn_kernel(const _Float16* __restrict__ qh, const _Float16* __restrict__ kh,
                            const _Float16* __restrict__ vh, const float* __restrict__ bias,
                            const unsigned char* __restrict__ mask,
                            float* __restrict__ attn_w, _Float16* __restrict__ ao) {
  extern __shared__ char smem[];
  float*    sc   = (float*)(smem + LDS_SC);
  _Float16* ps   = (_Float16*)(smem + LDS_PS);
  _Float16* kbuf0 = (_Float16*)(smem + LDS_KBUF0);
  _Float16* kbuf1 = (_Float16*)(smem + LDS_KBUF1);
  _Float16* vbuf0 = (_Float16*)(smem + LDS_VBUF0);
  _Float16* vbuf1 = (_Float16*)(smem + LDS_VBUF1);
  float*    rowm = (float*)(smem + LDS_STATS);
  float*    rowl = rowm + 16;

  const int qt = blockIdx.x % (SS / 16);
  const int h  = (blockIdx.x / (SS / 16)) % HH;
  const int b  = blockIdx.x / ((SS / 16) * HH);
  const int lane = threadIdx.x & 31;
  const int n = lane & 15;
  const int hi = lane >> 4;

  const size_t bh = (size_t)b * HH + h;
  const _Float16* Qp = qh + (bh * SS + (size_t)qt * 16) * HD;
  const v16h qa0 = load_frag_f16(Qp, HD);        // K = 0..31 of head dim
  const v16h qa1 = load_frag_f16(Qp + 32, HD);   // K = 32..63

  float rm[8], rl[8];
#pragma unroll
  for (int j = 0; j < 8; ++j) { rm[j] = -INFINITY; rl[j] = 0.f; }

  // ---- pass 1: score tiles, online max/sum, park raw scores in LDS --------
  const _Float16* Kbase = kh + bh * SS * HD;
#ifdef USE_TDM
  tdm_load_f16(LDS_KBUF0, Kbase, 16 * HD);       // prime the pipeline
#endif
  for (int kt = 0; kt <= qt; ++kt) {
    const _Float16* Kp;
#ifdef USE_TDM
    if (kt < qt) {
      tdm_load_f16(((kt + 1) & 1) ? LDS_KBUF1 : LDS_KBUF0,
                   Kbase + (size_t)(kt + 1) * 16 * HD, 16 * HD);
      __builtin_amdgcn_s_wait_tensorcnt(1);      // oldest (current tile) done
    } else {
      __builtin_amdgcn_s_wait_tensorcnt(0);
    }
    Kp = (kt & 1) ? kbuf1 : kbuf0;
#else
    Kp = Kbase + (size_t)kt * 16 * HD;
    __builtin_prefetch(Kp + 16 * HD, 0, 1);      // next K tile -> global_prefetch
#endif
    v16h kb0 = load_frag_f16(Kp, HD);
    v16h kb1 = load_frag_f16(Kp + 32, HD);
    v8f s = {};
    s = __builtin_amdgcn_wmma_f32_16x16x32_f16(false, qa0, false, kb0, (short)0, s,
                                               false, false);
    s = __builtin_amdgcn_wmma_f32_16x16x32_f16(false, qa1, false, kb1, (short)0, s,
                                               false, false);
    const int col = kt * 16 + n;
#pragma unroll
    for (int j = 0; j < 8; ++j) {
      const int row = qt * 16 + j + hi * 8;
      float v = s[j] + bias[((size_t)h * SS + row) * SS + col];
      if (mask[(size_t)row * SS + col]) v = -INFINITY;
      sc[(j + hi * 8) * SS + col] = v;
      // row max across the 16 lanes holding this row (xor masks stay in-group)
      float tmax = v;
#pragma unroll
      for (int m_ = 1; m_ < 16; m_ <<= 1) tmax = fmaxf(tmax, __shfl_xor(tmax, m_, 32));
      const float nm = fmaxf(rm[j], tmax);
      float ex = __expf(v - nm);
      float tsum = ex;
#pragma unroll
      for (int m_ = 1; m_ < 16; m_ <<= 1) tsum += __shfl_xor(tsum, m_, 32);
      rl[j] = rl[j] * __expf(rm[j] - nm) + tsum;
      rm[j] = nm;
    }
  }

  if (n == 0) {
#pragma unroll
    for (int j = 0; j < 8; ++j) { rowm[j + hi * 8] = rm[j]; rowl[j + hi * 8] = rl[j]; }
  }
  __syncthreads();

  // ---- pass 2: normalize, single write of attn weights, f16 P into LDS ----
  const int kmax = (qt + 1) * 16;
  const int kend = (kmax + 31) & ~31;            // pad to WMMA K granularity
  float* aw = attn_w + (bh * SS + (size_t)qt * 16) * SS;
  for (int r = 0; r < 16; ++r) {
    const float m = rowm[r];
    const float inv = 1.f / rowl[r];
    for (int c = lane; c < SS; c += 32) {
      float p = 0.f;
      if (c < kmax) p = __expf(sc[r * SS + c] - m) * inv;
      aw[(size_t)r * SS + c] = p;
      if (c < kend) ps[r * SS + c] = (_Float16)p;  // zero padding in [kmax,kend)
    }
  }
  __syncthreads();

  // ---- P @ V : out tile is 16 x 64 = four 16x16 accumulators --------------
  const _Float16* Vbase = vh + bh * SS * HD;
  v8f o0 = {}, o1 = {}, o2 = {}, o3 = {};
#ifdef USE_TDM
  tdm_load_f16(LDS_VBUF0, Vbase, 32 * HD);       // prime the pipeline
#endif
  for (int kc = 0; kc < kend; kc += 32) {
    const v16h pa = load_frag_f16(ps + kc, SS);  // A fragment from LDS
    const _Float16* Vp;
#ifdef USE_TDM
    if (kc + 32 < kend) {
      tdm_load_f16(((kc >> 5) + 1) & 1 ? LDS_VBUF1 : LDS_VBUF0,
                   Vbase + (size_t)(kc + 32) * HD, 32 * HD);
      __builtin_amdgcn_s_wait_tensorcnt(1);
    } else {
      __builtin_amdgcn_s_wait_tensorcnt(0);
    }
    Vp = ((kc >> 5) & 1) ? vbuf1 : vbuf0;
#else
    Vp = Vbase + (size_t)kc * HD;
#endif
    o0 = __builtin_amdgcn_wmma_f32_16x16x32_f16(false, pa, false,
            load_b_strided_f16(Vp + 0, HD), (short)0, o0, false, false);
    o1 = __builtin_amdgcn_wmma_f32_16x16x32_f16(false, pa, false,
            load_b_strided_f16(Vp + 16, HD), (short)0, o1, false, false);
    o2 = __builtin_amdgcn_wmma_f32_16x16x32_f16(false, pa, false,
            load_b_strided_f16(Vp + 32, HD), (short)0, o2, false, false);
    o3 = __builtin_amdgcn_wmma_f32_16x16x32_f16(false, pa, false,
            load_b_strided_f16(Vp + 48, HD), (short)0, o3, false, false);
  }

  // merged-heads f16 output [B, S, D] for the final projection
  const int col0 = h * HD;
#pragma unroll
  for (int j = 0; j < 8; ++j) {
    const int s = qt * 16 + j + hi * 8;
    _Float16* dst = ao + ((size_t)b * SS + s) * DD + col0 + n;
    dst[0]  = (_Float16)o0[j];
    dst[16] = (_Float16)o1[j];
    dst[32] = (_Float16)o2[j];
    dst[48] = (_Float16)o3[j];
  }
}

// ---------------------------------------------------------------------------
// Kernel 3: out = AO @ o_w^T + o_b  (f16 activations, f32 weights/out)
// ---------------------------------------------------------------------------
__global__ void proj_out_kernel(const _Float16* __restrict__ X, const float* __restrict__ W,
                                const float* __restrict__ bias, float* __restrict__ Y) {
  const int wave = blockIdx.x * (blockDim.x >> 5) + (threadIdx.x >> 5);
  const int nT = DD / 16;
  const int mt = wave / nT;
  const int nt = wave % nT;
  const int lane = threadIdx.x & 31;
  const _Float16* Xt = X + (size_t)mt * 16 * DD;
  const float* Wt = W + (size_t)nt * 16 * DD;
  v8f acc = {};
#pragma unroll 4
  for (int kt = 0; kt < DD; kt += 32) {
    v16h a = load_frag_f16(Xt + kt, DD);
    v16h b = load_frag_f32(Wt + kt, DD);
    acc = __builtin_amdgcn_wmma_f32_16x16x32_f16(false, a, false, b, (short)0, acc,
                                                 false, false);
  }
  const int hi = lane >> 4;
  const int n = nt * 16 + (lane & 15);
  const float bn = bias[n];
#pragma unroll
  for (int j = 0; j < 8; ++j) {
    const int row = mt * 16 + j + hi * 8;
    Y[(size_t)row * DD + n] = acc[j] + bn;
  }
}

// ---------------------------------------------------------------------------
extern "C" void kernel_launch(void* const* d_in, const int* in_sizes, int n_in,
                              void* d_out, int out_size, void* d_ws, size_t ws_size,
                              hipStream_t stream) {
  const float* q  = (const float*)d_in[0];
  const float* k  = (const float*)d_in[1];
  const float* v  = (const float*)d_in[2];
  const unsigned char* mask = (const unsigned char*)d_in[3];
  const float* bias = (const float*)d_in[4];
  const float* q_w = (const float*)d_in[5];
  const float* q_b = (const float*)d_in[6];
  const float* k_w = (const float*)d_in[7];
  const float* k_b = (const float*)d_in[8];
  const float* v_w = (const float*)d_in[9];
  const float* v_b = (const float*)d_in[10];
  const float* o_w = (const float*)d_in[11];
  const float* o_b = (const float*)d_in[12];

  float* out    = (float*)d_out;                       // [B, S, D]
  float* attn_w = out + (size_t)BB * SS * DD;          // [B, H, S, S]

  const size_t elems = (size_t)BB * SS * DD;
  _Float16* qh = (_Float16*)d_ws;        // [B, H, S, HD] f16
  _Float16* kh = qh + elems;
  _Float16* vh = kh + elems;
  _Float16* ao = vh + elems;             // [B, S, D] f16

  const int tiles = (BB * SS / 16) * (DD / 16);        // 32768 waves
  const dim3 blk(256);                                 // 8 waves / block
  const dim3 grd(tiles / 8);

  proj_qkv_kernel<<<grd, blk, 0, stream>>>(q, q_w, q_b, qh, 0.125f);  // HD^-0.5
  proj_qkv_kernel<<<grd, blk, 0, stream>>>(k, k_w, k_b, kh, 1.0f);
  proj_qkv_kernel<<<grd, blk, 0, stream>>>(v, v_w, v_b, vh, 1.0f);

  attn_kernel<<<dim3(BB * HH * (SS / 16)), dim3(32), LDS_TOTAL, stream>>>(
      qh, kh, vh, bias, mask, attn_w, ao);

  proj_out_kernel<<<grd, blk, 0, stream>>>(ao, o_w, o_b, out);
}
